// MultiHeadAttentionTorch_69647189672020
// MI455X (gfx1250) — compile-verified, matrix operand order checked by reading, not run on hardware
//
#include <hip/hip_runtime.h>
#include <hip/hip_bf16.h>

#define BATCH  4
#define SEQ    2048
#define DMODEL 1024
#define NHEADS 16
#define DHEAD  64
#define NEGVAL (-1000000000.0f)

typedef _Float16 half_t;
typedef __attribute__((ext_vector_type(16))) _Float16 v16h;
typedef __attribute__((ext_vector_type(8)))  _Float16 v8h;
typedef __attribute__((ext_vector_type(8)))  float    v8f;

// ---------------------------------------------------------------------------
// Fragment loader: 16 halves per lane for a 16x32 (MxK) A-tile or a (KxN)
// B-tile stored as 16 "major" rows x contiguous K.  Lane L<16 owns major
// index L with K = k0+0..7 and k0+16..23; lanes 16..31 own K = k0+8..15 and
// k0+24..31 (per CDNA5 16-bit WMMA VGPR layout).  Two 16-byte vector loads.
// ---------------------------------------------------------------------------
__device__ __forceinline__ v16h load_frag(const half_t* __restrict__ base,
                                          int ld, int k0, int lane) {
  int r  = lane & 15;
  int kb = (lane >> 4) << 3;              // 0 or 8
  const half_t* p = base + (size_t)r * ld + k0 + kb;
  v8h lo = *(const v8h*)(p);
  v8h hi = *(const v8h*)(p + 16);
  v16h f;
#pragma unroll
  for (int i = 0; i < 8; ++i) { f[i] = lo[i]; f[8 + i] = hi[i]; }
  return f;
}

__device__ __forceinline__ v8f wmma16(v16h a, v16h b, v8f c) {
  // D = A(16x32 f16) * B(32x16 f16) + C(16x16 f32)
  return __builtin_amdgcn_wmma_f32_16x16x32_f16(false, a, false, b,
                                                (short)0, c, false, false);
}

// ---------------------------------------------------------------------------
// f32 -> f16 conversion
// ---------------------------------------------------------------------------
__global__ void cvt_f32_f16(const float* __restrict__ src,
                            half_t* __restrict__ dst, int n) {
  int i = blockIdx.x * blockDim.x + threadIdx.x;
  if (i < n) dst[i] = (half_t)src[i];
}

// ---------------------------------------------------------------------------
// QKV projection: one wave computes a full 16(s) x 64(e) row-block for one
// (qkv, b, h, s-tile).  Per k-step: 1 A fragment (reused 4x) + 4 B fragments
// -> 4 WMMAs into 4 independent accumulators (no serialized D->C chain).
// q/k stored [b,h,s,e] f16; v stored transposed [b,h,e,s] f16.
// ---------------------------------------------------------------------------
__global__ void qkv_kernel(const half_t* __restrict__ xh,
                           const half_t* __restrict__ wqh,
                           const half_t* __restrict__ wkh,
                           const half_t* __restrict__ wvh,
                           half_t* __restrict__ qh,
                           half_t* __restrict__ kh,
                           half_t* __restrict__ vTh) {
  int wave = blockIdx.x * (blockDim.x >> 5) + (threadIdx.x >> 5);
  int lane = threadIdx.x & 31;
  int st  = wave & 127;          // seq tile   (128 x 16)
  int bh  = (wave >> 7) & 63;    // b*16+h
  int qkv = wave >> 13;          // 0=q 1=k 2=v  (waves = 3<<13)
  int h = bh & 15;

  const half_t* w = (qkv == 0) ? wqh : (qkv == 1) ? wkh : wvh;
  const half_t* A  = xh + ((size_t)(bh >> 4) * SEQ + (size_t)st * 16) * DMODEL;
  const half_t* Bp = w  + (size_t)h * DHEAD * DMODEL;

  v8f acc0 = {}, acc1 = {}, acc2 = {}, acc3 = {};
#pragma unroll 2
  for (int k = 0; k < DMODEL; k += 32) {
    v16h a  = load_frag(A, DMODEL, k, lane);
    v16h b0 = load_frag(Bp,                       DMODEL, k, lane);
    v16h b1 = load_frag(Bp + (size_t)16 * DMODEL, DMODEL, k, lane);
    v16h b2 = load_frag(Bp + (size_t)32 * DMODEL, DMODEL, k, lane);
    v16h b3 = load_frag(Bp + (size_t)48 * DMODEL, DMODEL, k, lane);
    acc0 = wmma16(a, b0, acc0);
    acc1 = wmma16(a, b1, acc1);
    acc2 = wmma16(a, b2, acc2);
    acc3 = wmma16(a, b3, acc3);
  }

  int col   = lane & 15;
  int rbase = (lane < 16) ? 0 : 8;
  float accs[4][8];
#pragma unroll
  for (int j = 0; j < 8; ++j) {
    accs[0][j] = acc0[j]; accs[1][j] = acc1[j];
    accs[2][j] = acc2[j]; accs[3][j] = acc3[j];
  }
  if (qkv < 2) {
    half_t* dst = ((qkv == 0) ? qh : kh) +
                  (((size_t)bh * SEQ + (size_t)st * 16) * DHEAD);
#pragma unroll
    for (int et = 0; et < 4; ++et)
#pragma unroll
      for (int j = 0; j < 8; ++j)
        dst[(size_t)(rbase + j) * DHEAD + et * 16 + col] = (half_t)accs[et][j];
  } else {
    half_t* dst = vTh + ((size_t)bh * DHEAD * SEQ) + st * 16;
#pragma unroll
    for (int et = 0; et < 4; ++et)
#pragma unroll
      for (int j = 0; j < 8; ++j)
        dst[(size_t)(et * 16 + col) * SEQ + rbase + j] = (half_t)accs[et][j];
  }
}

// ---------------------------------------------------------------------------
// Flash attention: one wave per (b,h,16 query rows).  Online softmax over
// causal key chunks of 32.  P is re-laid-out C-frag -> A-frag through LDS
// (wave-local; CDNA5 in-order DS + explicit s_wait_dscnt).
// ---------------------------------------------------------------------------
__global__ void attn_kernel(const half_t* __restrict__ qh,
                            const half_t* __restrict__ kh,
                            const half_t* __restrict__ vTh,
                            half_t* __restrict__ headsh) {
  __shared__ __align__(16) half_t pbuf[4][16][32];
  int wId  = threadIdx.x >> 5;
  int lane = threadIdx.x & 31;
  int wave = blockIdx.x * (blockDim.x >> 5) + wId;
  int qt = wave & 127;           // query tile
  int bh = wave >> 7;            // b*16+h
  int b = bh >> 4, h = bh & 15;
  int q0 = qt * 16;

  const half_t* Qbase = qh  + ((size_t)bh * SEQ + q0) * DHEAD;
  const half_t* Kbase = kh  + (size_t)bh * SEQ * DHEAD;
  const half_t* Vbase = vTh + (size_t)bh * DHEAD * SEQ;

  // Q fragments, e = 0..31 and 32..63 (reused across the whole key loop)
  v16h qa0 = load_frag(Qbase, DHEAD, 0,  lane);
  v16h qa1 = load_frag(Qbase, DHEAD, 32, lane);

  v8f o0 = {}, o1 = {}, o2 = {}, o3 = {};
  float mrow[8], lrow[8];
#pragma unroll
  for (int j = 0; j < 8; ++j) { mrow[j] = -1e30f; lrow[j] = 0.0f; }

  int col   = lane & 15;
  int rbase = (lane < 16) ? 0 : 8;

  for (int kk = 0; kk < q0 + 16; kk += 32) {
    // ----- scores S[16 x 32] = Q * K^T, two independent accumulators ------
    v8f s0 = {}, s1 = {};
    s0 = wmma16(qa0, load_frag(Kbase + (size_t)kk * DHEAD,        DHEAD, 0,  lane), s0);
    s1 = wmma16(qa0, load_frag(Kbase + (size_t)(kk + 16) * DHEAD, DHEAD, 0,  lane), s1);
    s0 = wmma16(qa1, load_frag(Kbase + (size_t)kk * DHEAD,        DHEAD, 32, lane), s0);
    s1 = wmma16(qa1, load_frag(Kbase + (size_t)(kk + 16) * DHEAD, DHEAD, 32, lane), s1);

    // ----- online softmax -------------------------------------------------
#pragma unroll
    for (int j = 0; j < 8; ++j) {
      int row = q0 + rbase + j;
      int t0 = kk + col, t1 = kk + 16 + col;
      float x0 = s0[j] * 0.125f; if (t0 > row) x0 = NEGVAL;
      float x1 = s1[j] * 0.125f; if (t1 > row) x1 = NEGVAL;
      // row max across the 16 lanes of this half-wave
      float m = fmaxf(x0, x1);
      m = fmaxf(m, __shfl_xor(m, 1, 32));
      m = fmaxf(m, __shfl_xor(m, 2, 32));
      m = fmaxf(m, __shfl_xor(m, 4, 32));
      m = fmaxf(m, __shfl_xor(m, 8, 32));
      float newm = fmaxf(mrow[j], m);
      float corr = __expf(mrow[j] - newm);
      float p0 = __expf(x0 - newm);
      float p1 = __expf(x1 - newm);
      float ps = p0 + p1;
      ps += __shfl_xor(ps, 1, 32);
      ps += __shfl_xor(ps, 2, 32);
      ps += __shfl_xor(ps, 4, 32);
      ps += __shfl_xor(ps, 8, 32);
      lrow[j] = lrow[j] * corr + ps;
      mrow[j] = newm;
      o0[j] *= corr; o1[j] *= corr; o2[j] *= corr; o3[j] *= corr;
      pbuf[wId][rbase + j][col]      = (half_t)p0;
      pbuf[wId][rbase + j][16 + col] = (half_t)p1;
    }

    // wave-local: DS ops are in-order; wait for stores, fence the compiler
    asm volatile("s_wait_dscnt 0x0" ::: "memory");

    // ----- O += P * V : 1 A fragment reused by 4 independent WMMAs --------
    v16h pa = load_frag(&pbuf[wId][0][0], 32, 0, lane);
    o0 = wmma16(pa, load_frag(Vbase,                    SEQ, kk, lane), o0);
    o1 = wmma16(pa, load_frag(Vbase + (size_t)16 * SEQ, SEQ, kk, lane), o1);
    o2 = wmma16(pa, load_frag(Vbase + (size_t)32 * SEQ, SEQ, kk, lane), o2);
    o3 = wmma16(pa, load_frag(Vbase + (size_t)48 * SEQ, SEQ, kk, lane), o3);
    asm volatile("" ::: "memory");  // keep next iter's DS stores after these loads
  }

  // ----- normalize and store heads: [b, s, h*64] f16 ----------------------
  half_t* dst = headsh + (((size_t)b * SEQ + q0) * (NHEADS * DHEAD)) + h * DHEAD;
#pragma unroll
  for (int j = 0; j < 8; ++j) {
    float inv = 1.0f / lrow[j];
    size_t ro = (size_t)(rbase + j) * (NHEADS * DHEAD);
    dst[ro + col]      = (half_t)(o0[j] * inv);
    dst[ro + 16 + col] = (half_t)(o1[j] * inv);
    dst[ro + 32 + col] = (half_t)(o2[j] * inv);
    dst[ro + 48 + col] = (half_t)(o3[j] * inv);
  }
}

// ---------------------------------------------------------------------------
// Output projection: out[r, m] = sum_k heads[r, k] * wo[m, k]  (f32 out).
// One wave computes 16 rows x 64 output cols: 1 A frag + 4 B frags -> 4 WMMA.
// ---------------------------------------------------------------------------
__global__ void outproj_kernel(const half_t* __restrict__ headsh,
                               const half_t* __restrict__ woh,
                               float* __restrict__ out) {
  int wave = blockIdx.x * (blockDim.x >> 5) + (threadIdx.x >> 5);
  int lane = threadIdx.x & 31;
  int ct = wave & 15;      // 16 column tiles of 64
  int rt = wave >> 4;      // 512 row tiles of 16  (rows = b*s)

  const half_t* A  = headsh + (size_t)rt * 16 * DMODEL;
  const half_t* Bp = woh    + (size_t)ct * 64 * DMODEL;   // wo[m][k], k contiguous

  v8f acc0 = {}, acc1 = {}, acc2 = {}, acc3 = {};
#pragma unroll 2
  for (int k = 0; k < DMODEL; k += 32) {
    v16h a  = load_frag(A, DMODEL, k, lane);
    v16h b0 = load_frag(Bp,                       DMODEL, k, lane);
    v16h b1 = load_frag(Bp + (size_t)16 * DMODEL, DMODEL, k, lane);
    v16h b2 = load_frag(Bp + (size_t)32 * DMODEL, DMODEL, k, lane);
    v16h b3 = load_frag(Bp + (size_t)48 * DMODEL, DMODEL, k, lane);
    acc0 = wmma16(a, b0, acc0);
    acc1 = wmma16(a, b1, acc1);
    acc2 = wmma16(a, b2, acc2);
    acc3 = wmma16(a, b3, acc3);
  }

  int col   = lane & 15;
  int rbase = (lane < 16) ? 0 : 8;
  float* dst = out + (size_t)rt * 16 * DMODEL + ct * 64;
#pragma unroll
  for (int j = 0; j < 8; ++j) {
    size_t ro = (size_t)(rbase + j) * DMODEL;
    dst[ro + col]      = acc0[j];
    dst[ro + 16 + col] = acc1[j];
    dst[ro + 32 + col] = acc2[j];
    dst[ro + 48 + col] = acc3[j];
  }
}

// ---------------------------------------------------------------------------
extern "C" void kernel_launch(void* const* d_in, const int* in_sizes, int n_in,
                              void* d_out, int out_size, void* d_ws, size_t ws_size,
                              hipStream_t stream) {
  const float* x  = (const float*)d_in[0];
  const float* wq = (const float*)d_in[1];
  const float* wk = (const float*)d_in[2];
  const float* wv = (const float*)d_in[3];
  const float* wo = (const float*)d_in[4];
  float* out = (float*)d_out;

  // workspace layout (halves)
  const size_t NX  = (size_t)BATCH * SEQ * DMODEL;          // 8388608
  const size_t NW  = (size_t)NHEADS * DHEAD * DMODEL;       // 1048576
  const size_t NWO = (size_t)DMODEL * DMODEL;               // 1048576
  const size_t NQ  = (size_t)BATCH * NHEADS * SEQ * DHEAD;  // 8388608

  half_t* ws  = (half_t*)d_ws;
  half_t* xh  = ws;
  half_t* wqh = xh  + NX;
  half_t* wkh = wqh + NW;
  half_t* wvh = wkh + NW;
  half_t* woh = wvh + NW;
  half_t* qh  = woh + NWO;
  half_t* kh  = qh  + NQ;
  half_t* vTh = kh  + NQ;
  half_t* hds = vTh + NQ;

  // 1) precision conversion
  cvt_f32_f16<<<(int)((NX  + 255) / 256), 256, 0, stream>>>(x,  xh,  (int)NX);
  cvt_f32_f16<<<(int)((NW  + 255) / 256), 256, 0, stream>>>(wq, wqh, (int)NW);
  cvt_f32_f16<<<(int)((NW  + 255) / 256), 256, 0, stream>>>(wk, wkh, (int)NW);
  cvt_f32_f16<<<(int)((NW  + 255) / 256), 256, 0, stream>>>(wv, wvh, (int)NW);
  cvt_f32_f16<<<(int)((NWO + 255) / 256), 256, 0, stream>>>(wo, woh, (int)NWO);

  // 2) QKV projections: 3 * 64 * 128 = 24576 waves (16x64 tile each)
  qkv_kernel<<<24576 / 4, 128, 0, stream>>>(xh, wqh, wkh, wvh, qh, kh, vTh);

  // 3) flash attention: 64 * 128 = 8192 waves
  attn_kernel<<<8192 / 4, 128, 0, stream>>>(qh, kh, vTh, hds);

  // 4) output projection: 512 * 16 = 8192 waves (16x64 tile each)
  outproj_kernel<<<8192 / 4, 128, 0, stream>>>(hds, woh, out);
}
